// DynamicMaskHead_33878702031551
// MI455X (gfx1250) — matrix operand order repaired and checked by reading, not compiled, not run on hardware
//
#include <hip/hip_runtime.h>
#include <hip/hip_bf16.h>

typedef __attribute__((ext_vector_type(16))) _Float16 v16h;
typedef __attribute__((ext_vector_type(8)))  float    v8f;

#define NH 136
#define NW 200
#define NHW 27200          // NH*NW
#define OH 272
#define OW 400
#define OHW 108800         // OH*OW
#define NPARAM 169
#define NTILES 1700        // NHW / 16
#define NWAVES 8

// LDS layout (dynamic):
//   float logits[NHW]    108800 B
//   float red[3*256]       3072 B
#define SMEM_BYTES (NHW*4 + 3*256*4)

__global__ __launch_bounds__(256)
void dyn_mask_head_kernel(const float* __restrict__ mask_feats,
                          const float* __restrict__ params,
                          const float* __restrict__ inst_loc,
                          const float* __restrict__ gt_masks,
                          const int*   __restrict__ im_inds,
                          const int*   __restrict__ fpn_levels,
                          const int*   __restrict__ stride_p,
                          float*       __restrict__ out)
{
    extern __shared__ char smem_raw[];
    float* logits = (float*)smem_raw;
    float* red    = logits + NHW;

    const int inst = blockIdx.x;
    const int tid  = threadIdx.x;
    const int L    = tid & 31;       // lane in wave32
    const int wv   = tid >> 5;       // wave id (0..7)

    const float* P = params + (size_t)inst * NPARAM;

    const int   img     = im_inds[inst];
    const int   lvl     = fpn_levels[inst];
    const float inv_soi = 1.0f / (float)(64 << lvl);     // SOI = 64 * 2^lvl
    const float ix      = inst_loc[inst * 2 + 0];
    const float iy      = inst_loc[inst * 2 + 1];
    const int   stride  = stride_p[0];
    const float halfstr = (float)(stride >> 1);
    const float* F = mask_feats + (size_t)img * (8 * NHW);

    // ---- uniform scalars: biases, w2, b2 (scalar loads, uniform address) ----
    float b0r[8], b1r[8], w2r[8];
    #pragma unroll
    for (int e = 0; e < 8; ++e) {
        b0r[e] = P[152 + e];
        b1r[e] = P[160 + e];
        w2r[e] = P[144 + e];
    }
    const float b2v = P[168];

    // ---- constant A fragments: W0 (16x32 f16) and W1 (16x32 f16) ----
    // A layout: lane<16 row M=L: elt e -> K=e (K=0..15, same order as B);
    //           lane>=16 row M=L-16: elt e -> K=16+e (K=16..31).
    // W0: rows n=0..7, K = input channel 0..9. W1: rows n=0..7, K=0..7.
    v16h w0f = {};
    v16h w1f = {};
    if (L < 8) {
        #pragma unroll
        for (int k = 0; k < 10; ++k) w0f[k] = (_Float16)P[L * 10 + k];      // K=0..9
        #pragma unroll
        for (int k = 0; k < 8; ++k) w1f[k] = (_Float16)P[80 + L * 8 + k];   // K=0..7
    } else if (L >= 16 && L < 24) {
        const int r = L - 16;
        w0f[0] = (_Float16)P[r * 10 + 8];   // K=16 -> feat6 weight, meets zero B
        w0f[1] = (_Float16)P[r * 10 + 9];   // K=17 -> feat7 weight, meets zero B
    }

    const int m = L & 15;   // pixel column within tile

    // ---- per-tile MLP, fully register-resident between layers ----
    for (int tile = wv; tile < NTILES; tile += NWAVES) {
        const int p = tile * 16 + m;

        // B fragment = X^T (32x16): lane<16 holds column (pixel) m,
        // elements e -> K=e: 0=relx, 1=rely, 2..9=feat0..7, 10..15=0.
        // lanes>=16 hold K=16..31 -> all zero.
        v16h xb = {};
        if (L < 16) {
            const int py = p / NW;
            const int px = p - py * NW;
            const float relx = (ix - ((float)(px * stride) + halfstr)) * inv_soi;
            const float rely = (iy - ((float)(py * stride) + halfstr)) * inv_soi;
            xb[0] = (_Float16)relx;
            xb[1] = (_Float16)rely;
            #pragma unroll
            for (int c = 0; c < 8; ++c) xb[2 + c] = (_Float16)F[c * NHW + p];
        }

        // layer 0: D0^T = W0 * X^T  -> channel j in VGPR j, pixel in lane
        v8f d0 = {};
        d0 = __builtin_amdgcn_wmma_f32_16x16x32_f16(
                 false, w0f, false, xb, (short)0, d0, false, false);

        // bias + relu, lane-local repack into layer-1 B fragment
        v16h hb = {};
        if (L < 16) {
            #pragma unroll
            for (int e = 0; e < 8; ++e) {
                float h = d0[e] + b0r[e];
                h = h > 0.0f ? h : 0.0f;
                hb[e] = (_Float16)h;
            }
        }

        // layer 1: D1^T = W1 * H1^T
        v8f d1 = {};
        d1 = __builtin_amdgcn_wmma_f32_16x16x32_f16(
                 false, w1f, false, hb, (short)0, d1, false, false);

        // layer 2: lane-local dot with w2, write logit
        if (L < 16) {
            float lg = b2v;
            #pragma unroll
            for (int e = 0; e < 8; ++e) {
                float h = d1[e] + b1r[e];
                h = h > 0.0f ? h : 0.0f;
                lg = fmaf(w2r[e], h, lg);
            }
            logits[p] = lg;
        }
    }

    __syncthreads();

    // ---- bilinear x2 upsample (align_corners) + sigmoid + dice partials ----
    const float* gt = gt_masks + (size_t)inst * OHW;
    const float sy = 135.0f / 271.0f;   // (H-1)/(OH-1)
    const float sx = 199.0f / 399.0f;   // (W-1)/(OW-1)
    float accI = 0.0f, accS = 0.0f, accT = 0.0f;

    for (int idx = tid; idx < OHW; idx += 256) {
        const int oy = idx / OW;
        const int ox = idx - oy * OW;
        const float fy = (float)oy * sy;
        const float fx = (float)ox * sx;
        int y0 = (int)fy, x0 = (int)fx;
        int y1 = y0 + 1; if (y1 > NH - 1) y1 = NH - 1;
        int x1 = x0 + 1; if (x1 > NW - 1) x1 = NW - 1;
        const float wy = fy - (float)y0;
        const float wx = fx - (float)x0;

        const float* r0 = logits + y0 * NW;
        const float* r1 = logits + y1 * NW;
        const float top = r0[x0] * (1.0f - wx) + r0[x1] * wx;
        const float bot = r1[x0] * (1.0f - wx) + r1[x1] * wx;
        const float v   = top * (1.0f - wy) + bot * wy;

        const float s = 1.0f / (1.0f + __expf(-v));
        const float t = gt[idx];
        accI += s * t;
        accS += s * s;
        accT += t * t;
    }

    red[tid]       = accI;
    red[256 + tid] = accS;
    red[512 + tid] = accT;
    __syncthreads();

    for (int off = 128; off > 0; off >>= 1) {
        if (tid < off) {
            red[tid]       += red[tid + off];
            red[256 + tid] += red[256 + tid + off];
            red[512 + tid] += red[512 + tid + off];
        }
        __syncthreads();
    }

    if (tid == 0) {
        const float I = red[0];
        const float U = red[256] + red[512] + 1e-5f;
        out[inst] = 1.0f - 2.0f * I / U;
    }
}

extern "C" void kernel_launch(void* const* d_in, const int* in_sizes, int n_in,
                              void* d_out, int out_size, void* d_ws, size_t ws_size,
                              hipStream_t stream) {
    (void)in_sizes; (void)n_in; (void)out_size; (void)d_ws; (void)ws_size;

    const float* mask_feats = (const float*)d_in[0];
    const float* params     = (const float*)d_in[1];
    const float* inst_loc   = (const float*)d_in[2];
    const float* gt_masks   = (const float*)d_in[3];
    const int*   im_inds    = (const int*)d_in[4];
    const int*   fpn_levels = (const int*)d_in[5];
    const int*   stride_p   = (const int*)d_in[6];
    float*       out        = (float*)d_out;

    dyn_mask_head_kernel<<<512, 256, SMEM_BYTES, stream>>>(
        mask_feats, params, inst_loc, gt_masks, im_inds, fpn_levels, stride_p, out);
}
